// GCN_52364241273249
// MI455X (gfx1250) — matrix operand order repaired and checked by reading, not compiled, not run on hardware
//
#include <hip/hip_runtime.h>
#include <hip/hip_bf16.h>

// ---------------------------------------------------------------------------
// Two-layer sampled-GCN, MI455X (gfx1250, wave32, WMMA).
//
// Exact structural optimization (see gcn_norm semantics in the reference):
//   layer 0: norm==0 for every edge with row >= N1  -> H only needed for rows < N1
//   layer 1: norm==0 for every edge with row >= N2  -> layer-0 output only needed
//            for rows < N2
// Pipeline (all on `stream`, ws-resident intermediates):
//   1) zero deg0/deg1/out0
//   2) deg0 = scatter-count(col0);  deg1 = scatter-count(col1)
//   3) pack  x[0:N1,0:100] -> bf16 [N1,128] (K zero-padded)
//      pack  W0^T           -> bf16 [256,128]
//   4) H = Xb @ W0b   (v_wmma_f32_16x16x32_bf16, fp32 accum) -> f32 [N1,256]
//   5) scatter layer 0: out0[c] += dis(r)*dis(c)*H[r]   (only r<N1 && c<N2)
//   6) h1b = bf16(relu(out0 + b0))  [N2,256];  pack W1^T -> bf16 [48,256]
//   7) G = h1b @ W1b  (WMMA) -> f32 [N2,48]
//   8) out = b1 (broadcast);  scatter layer 1: out[c] += dis1(r)*dis1(c)*G[r,0:47]
// ---------------------------------------------------------------------------

typedef __attribute__((ext_vector_type(16))) __bf16 v16bf;
typedef __attribute__((ext_vector_type(8)))  __bf16 v8bf;
typedef __attribute__((ext_vector_type(8)))  float  v8f;

constexpr int N0 = 409600, N1 = 40960, N2 = 4096;
constexpr int D_IN = 100, D_H = 256, D_OUT = 47;
constexpr int KP0 = 128;   // layer-0 GEMM K padded 100 -> 128
constexpr int NP1 = 48;    // layer-1 GEMM N padded 47 -> 48

constexpr size_t alignup(size_t x) { return (x + 255) & ~(size_t)255; }
constexpr size_t OFF_DEG0 = 0;                                              // f32[N1]
constexpr size_t OFF_DEG1 = alignup(OFF_DEG0 + (size_t)N1 * 4);             // f32[N2]
constexpr size_t OFF_XB   = alignup(OFF_DEG1 + (size_t)N2 * 4);             // bf16[N1][KP0]
constexpr size_t OFF_W0T  = alignup(OFF_XB  + (size_t)N1 * KP0 * 2);        // bf16[D_H][KP0]
constexpr size_t OFF_H    = alignup(OFF_W0T + (size_t)D_H * KP0 * 2);       // f32[N1][D_H]
constexpr size_t OFF_OUT0 = alignup(OFF_H   + (size_t)N1 * D_H * 4);        // f32[N2][D_H]
constexpr size_t OFF_H1B  = alignup(OFF_OUT0 + (size_t)N2 * D_H * 4);       // bf16[N2][D_H]
constexpr size_t OFF_W1T  = alignup(OFF_H1B + (size_t)N2 * D_H * 2);        // bf16[NP1][D_H]
constexpr size_t OFF_G    = alignup(OFF_W1T + (size_t)NP1 * D_H * 2);       // f32[N2][NP1]

// --------------------------------- kernels ---------------------------------

__global__ void zero_f32(float* __restrict__ p, int n) {
  for (int i = blockIdx.x * blockDim.x + threadIdx.x; i < n;
       i += gridDim.x * blockDim.x)
    p[i] = 0.0f;
}

__global__ void count_deg(const int* __restrict__ col, int E,
                          float* __restrict__ deg) {
  for (int e = blockIdx.x * blockDim.x + threadIdx.x; e < E;
       e += gridDim.x * blockDim.x)
    atomicAdd(&deg[col[e]], 1.0f);
}

__global__ void pack_x_bf16(const float* __restrict__ x, __bf16* __restrict__ xb,
                            int rows, int dIn, int kPad) {
  int i = blockIdx.x * blockDim.x + threadIdx.x;
  if (i >= rows * kPad) return;
  int r = i / kPad, k = i - r * kPad;
  float v = (k < dIn) ? x[(size_t)r * dIn + k] : 0.0f;
  xb[i] = (__bf16)v;
}

// Wt[n][k] = W[k][n]  (zero-padded in both K and N)
__global__ void pack_wT_bf16(const float* __restrict__ W, __bf16* __restrict__ Wt,
                             int K, int N, int kPad, int nPad) {
  int i = blockIdx.x * blockDim.x + threadIdx.x;
  if (i >= nPad * kPad) return;
  int n = i / kPad, k = i - n * kPad;
  float v = (k < K && n < N) ? W[(size_t)k * N + n] : 0.0f;
  Wt[i] = (__bf16)v;
}

// One wave per 16x16 output tile; Bt is N-major (K contiguous per column).
// A-lane layout / B-lane layout / D-lane layout per CDNA5 ISA 7.12.2.
template <int KSTEPS>
__global__ void gemm_wmma_bf16(const __bf16* __restrict__ A, int lda,
                               const __bf16* __restrict__ Bt, int ldb,
                               float* __restrict__ C, int ldc,
                               int Mtiles, int Ntiles) {
  const int wave = blockIdx.x * (blockDim.x >> 5) + (threadIdx.x >> 5);
  const int lane = threadIdx.x & 31;
  const int tM = wave / Ntiles;
  const int tN = wave - tM * Ntiles;
  if (tM >= Mtiles) return;              // wave-uniform: EXEC stays all-ones
  const int hh  = lane >> 4;             // lane half (0: lanes 0-15, 1: 16-31)
  const int idx = lane & 15;
  const __bf16* Arow = A  + (size_t)(tM * 16 + idx) * lda;
  const __bf16* Bcol = Bt + (size_t)(tN * 16 + idx) * ldb;
  v8f acc = {};
#pragma unroll
  for (int s = 0; s < KSTEPS; ++s) {
    const int k0 = s * 32;
    // A row M=idx: halfs hold K = {8h..8h+7} and {16+8h..16+8h+7}
    v8bf alo = *(const v8bf*)(Arow + k0 + 8 * hh);
    v8bf ahi = *(const v8bf*)(Arow + k0 + 16 + 8 * hh);
    v16bf a;
#pragma unroll
    for (int i = 0; i < 8; ++i) { a[i] = alo[i]; a[i + 8] = ahi[i]; }
    // B col N=idx: halfs hold K = {8h..8h+7} and {16+8h..16+8h+7}
    v8bf blo = *(const v8bf*)(Bcol + k0 + 8 * hh);
    v8bf bhi = *(const v8bf*)(Bcol + k0 + 16 + 8 * hh);
    v16bf b;
#pragma unroll
    for (int i = 0; i < 8; ++i) { b[i] = blo[i]; b[i + 8] = bhi[i]; }
    acc = __builtin_amdgcn_wmma_f32_16x16x32_bf16(
        false, a, false, b, (short)0, acc, false, false);
  }
  // D: element v -> (M = v + 8h, N = idx)
  float* Cout = C + (size_t)(tM * 16 + 8 * hh) * ldc + tN * 16 + idx;
#pragma unroll
  for (int v = 0; v < 8; ++v) Cout[(size_t)v * ldc] = acc[v];
}

// one wave per edge; only edges with r<N1 (nonzero norm) and c<N2 (live dst)
__global__ void scatter_l0(const int* __restrict__ row, const int* __restrict__ col,
                           int E, const float* __restrict__ deg,
                           const float* __restrict__ H, float* __restrict__ out0) {
  int e = blockIdx.x * (blockDim.x >> 5) + (threadIdx.x >> 5);
  int lane = threadIdx.x & 31;
  if (e >= E) return;
  int r = row[e], c = col[e];
  if (r >= N1 || c >= N2) return;        // norm == 0 or dst not consumed downstream
  float dr = deg[r], dc = deg[c];
  if (dr <= 0.0f || dc <= 0.0f) return;  // dis == 0
  float nrm = rsqrtf(fmaxf(dr, 1.0f)) * rsqrtf(fmaxf(dc, 1.0f));
  const float* h = H + (size_t)r * D_H;
  float* o = out0 + (size_t)c * D_H;
#pragma unroll
  for (int j = lane; j < D_H; j += 32) atomicAdd(o + j, nrm * h[j]);
}

__global__ void bias_relu_pack(const float* __restrict__ out0,
                               const float* __restrict__ b0,
                               __bf16* __restrict__ h1b, int rows, int d) {
  int i = blockIdx.x * blockDim.x + threadIdx.x;
  if (i >= rows * d) return;
  int k = i % d;
  float v = out0[i] + b0[k];
  h1b[i] = (__bf16)fmaxf(v, 0.0f);
}

__global__ void init_out_bias(float* __restrict__ out, const float* __restrict__ b1,
                              int rows, int d) {
  int i = blockIdx.x * blockDim.x + threadIdx.x;
  if (i >= rows * d) return;
  out[i] = b1[i % d];
}

__global__ void scatter_l1(const int* __restrict__ row, const int* __restrict__ col,
                           int E, const float* __restrict__ deg1,
                           const float* __restrict__ G, float* __restrict__ out) {
  int e = blockIdx.x * (blockDim.x >> 5) + (threadIdx.x >> 5);
  int lane = threadIdx.x & 31;
  if (e >= E) return;
  int r = row[e], c = col[e];
  if (r >= N2 || c >= N2) return;        // norm == 0 for r >= N2
  float dr = deg1[r], dc = deg1[c];
  if (dr <= 0.0f || dc <= 0.0f) return;
  float nrm = rsqrtf(fmaxf(dr, 1.0f)) * rsqrtf(fmaxf(dc, 1.0f));
  const float* g = G + (size_t)r * NP1;
  float* o = out + (size_t)c * D_OUT;
  for (int j = lane; j < D_OUT; j += 32) atomicAdd(o + j, nrm * g[j]);
}

// ------------------------------ host launcher ------------------------------

extern "C" void kernel_launch(void* const* d_in, const int* in_sizes, int n_in,
                              void* d_out, int out_size, void* d_ws, size_t ws_size,
                              hipStream_t stream) {
  (void)n_in; (void)out_size; (void)ws_size;
  const float* x   = (const float*)d_in[0];
  const int*   ei0 = (const int*)d_in[1];
  const int*   ei1 = (const int*)d_in[2];
  const float* W0  = (const float*)d_in[3];
  const float* b0  = (const float*)d_in[4];
  const float* W1  = (const float*)d_in[5];
  const float* b1  = (const float*)d_in[6];
  float* out = (float*)d_out;

  const int E0t = in_sizes[1] / 2;   // 614400 + 40960
  const int E1t = in_sizes[2] / 2;   // 61440 + 4096
  const int* row0 = ei0;             const int* col0 = ei0 + E0t;
  const int* row1 = ei1;             const int* col1 = ei1 + E1t;

  char* ws = (char*)d_ws;
  float*  deg0 = (float*)(ws + OFF_DEG0);
  float*  deg1 = (float*)(ws + OFF_DEG1);
  __bf16* Xb   = (__bf16*)(ws + OFF_XB);
  __bf16* W0T  = (__bf16*)(ws + OFF_W0T);
  float*  H    = (float*)(ws + OFF_H);
  float*  out0 = (float*)(ws + OFF_OUT0);
  __bf16* h1b  = (__bf16*)(ws + OFF_H1B);
  __bf16* W1T  = (__bf16*)(ws + OFF_W1T);
  float*  G    = (float*)(ws + OFF_G);

  const int T = 256;
  auto blk = [](long long n, int t) { return (int)((n + t - 1) / t); };

  // 1) zero accumulators
  zero_f32<<<blk(N1 + N2, T), T, 0, stream>>>(deg0, N1 + N2 /* deg0|deg1 contig-ish */);
  zero_f32<<<blk((long long)N2 * D_H, T), T, 0, stream>>>(out0, N2 * D_H);
  // deg1 may not be adjacent to deg0 after alignment; zero it explicitly too
  zero_f32<<<blk(N2, T), T, 0, stream>>>(deg1, N2);

  // 2) degrees
  count_deg<<<blk(E0t, T), T, 0, stream>>>(col0, E0t, deg0);
  count_deg<<<blk(E1t, T), T, 0, stream>>>(col1, E1t, deg1);

  // 3) pack bf16 operands (K zero-padded)
  pack_x_bf16<<<blk((long long)N1 * KP0, T), T, 0, stream>>>(x, Xb, N1, D_IN, KP0);
  pack_wT_bf16<<<blk((long long)D_H * KP0, T), T, 0, stream>>>(W0, W0T, D_IN, D_H, KP0, D_H);

  // 4) H[N1,256] = Xb @ W0  (WMMA bf16 -> f32)
  {
    int Mt = N1 / 16, Nt = D_H / 16;                 // 2560 x 16 tiles
    int waves = Mt * Nt;
    gemm_wmma_bf16<KP0 / 32><<<blk((long long)waves * 32, T), T, 0, stream>>>(
        Xb, KP0, W0T, KP0, H, D_H, Mt, Nt);
  }

  // 5) layer-0 gather/scale/scatter into out0[N2,256]
  scatter_l0<<<blk((long long)E0t * 32, T), T, 0, stream>>>(row0, col0, E0t, deg0, H, out0);

  // 6) h1 = relu(out0 + b0) packed bf16; pack W1^T
  bias_relu_pack<<<blk((long long)N2 * D_H, T), T, 0, stream>>>(out0, b0, h1b, N2, D_H);
  pack_wT_bf16<<<blk((long long)NP1 * D_H, T), T, 0, stream>>>(W1, W1T, D_H, D_OUT, D_H, NP1);

  // 7) G[N2,48] = h1b @ W1  (WMMA)
  {
    int Mt = N2 / 16, Nt = NP1 / 16;                 // 256 x 3 tiles
    int waves = Mt * Nt;
    gemm_wmma_bf16<D_H / 32><<<blk((long long)waves * 32, T), T, 0, stream>>>(
        h1b, D_H, W1T, D_H, G, NP1, Mt, Nt);
  }

  // 8) out = b1 + layer-1 scatter
  init_out_bias<<<blk((long long)N2 * D_OUT, T), T, 0, stream>>>(out, b1, N2, D_OUT);
  scatter_l1<<<blk((long long)E1t * 32, T), T, 0, stream>>>(row1, col1, E1t, deg1, G, out);
}
